// ReformerEncoderLayer_9483287790117
// MI455X (gfx1250) — matrix operand order repaired and checked by reading, not compiled, not run on hardware
//
#include <hip/hip_runtime.h>

typedef __bf16 bf16;
typedef __attribute__((ext_vector_type(16))) __bf16 v16bf;
typedef __attribute__((ext_vector_type(8)))  __bf16 v8bf;
typedef __attribute__((ext_vector_type(8)))  float  v8f;

#define BB   8
#define TT   4096
#define DD   512
#define FF   2048
#define HH   8
#define DH   64
#define BT   (BB * TT)          // 32768 rows
#define NCH  (TT / 32)          // 128 chunks per head-sequence
#define NB2  64                 // n_buckets/2

// ---------------------------------------------------------------------------
// Weight prep: f32 (KxN) -> bf16 transposed (NxK). One-time, tiny.
// ---------------------------------------------------------------------------
__global__ void cvt_transpose_bf16(const float* __restrict__ in, bf16* __restrict__ out,
                                   int K, int N) {
    for (int i = blockIdx.x * blockDim.x + threadIdx.x; i < K * N;
         i += gridDim.x * blockDim.x) {
        int k = i / N, n = i % N;
        out[(size_t)n * K + k] = (bf16)in[i];
    }
}

// ---------------------------------------------------------------------------
// LayerNorm over D=512, output bf16. One block per row, 256 threads.
// ---------------------------------------------------------------------------
__global__ __launch_bounds__(256)
void ln_to_bf16(const float* __restrict__ x, const float* __restrict__ g,
                const float* __restrict__ b, bf16* __restrict__ out) {
    const int row = blockIdx.x;
    const int tid = threadIdx.x;
    const float* xr = x + (size_t)row * DD;
    float a0 = xr[tid], a1 = xr[tid + 256];
    __shared__ float red[256];
    red[tid] = a0 + a1;
    __syncthreads();
    for (int s = 128; s > 0; s >>= 1) { if (tid < s) red[tid] += red[tid + s]; __syncthreads(); }
    float mean = red[0] * (1.f / DD);
    __syncthreads();
    float d0 = a0 - mean, d1 = a1 - mean;
    red[tid] = d0 * d0 + d1 * d1;
    __syncthreads();
    for (int s = 128; s > 0; s >>= 1) { if (tid < s) red[tid] += red[tid + s]; __syncthreads(); }
    float rstd = rsqrtf(red[0] * (1.f / DD) + 1e-5f);
    bf16* orow = out + (size_t)row * DD;
    orow[tid]       = (bf16)(d0 * rstd * g[tid]       + b[tid]);
    orow[tid + 256] = (bf16)(d1 * rstd * g[tid + 256] + b[tid + 256]);
}

// ---------------------------------------------------------------------------
// bf16 WMMA GEMM: C(MxN) = A(MxK) * Bt^T, with Bt given N-major (NxK).
// Block: 256 threads (8 waves), tile 128x128, BK=32.
// Wave -> 32x64 output (2x4 frags) => 8 v_wmma per K-step per wave.
// Both tiles filled with GLOBAL_LOAD_ASYNC_TO_LDS_B128; triple-buffered
// software pipeline on ASYNCcnt (tile t+2 in flight during compute of t).
// EPI 0: out bf16
// EPI 1: outF = res1 + acc + bias                       (y1 = x + proj)
// EPI 2: outB = bf16(relu(acc + bias))                  (FFN mid)
// EPI 3: outF = 0.5*(res1 + res2 + acc + bias)          (final average)
// ---------------------------------------------------------------------------
template <int EPI>
__global__ __launch_bounds__(256)
void gemm_bf16(const bf16* __restrict__ A, const bf16* __restrict__ Bt,
               int M, int N, int K,
               float* __restrict__ outF, bf16* __restrict__ outB,
               const float* __restrict__ bias,
               const float* __restrict__ res1, const float* __restrict__ res2) {
    __shared__ __align__(16) bf16 As[3][128][40];   // 128(M) x 32(K) x3 bufs
    __shared__ __align__(16) bf16 Bs[3][128][40];   // 128(N) x 32(K) x3 bufs

    const int tid  = threadIdx.x;
    const int lane = tid & 31;
    const int wave = tid >> 5;
    const int half = lane >> 4;          // lane 0-15 vs 16-31
    const int l16  = lane & 15;
    const int wm   = (wave >> 1) * 32;   // wave row offset in block tile
    const int wn   = (wave & 1) * 64;    // wave col offset in block tile
    const int mBlock = blockIdx.y * 128;
    const int nBlock = blockIdx.x * 128;
    const int nT = K >> 5;               // K/32 steps (>= 16 here)

    // per-thread fill coordinates: 512 chunks of 8 bf16, 2 per thread
    const int r0 = tid >> 2, c0 = (tid & 3) << 3;          // chunk tid
    const int r1 = (tid + 256) >> 2, c1 = ((tid + 256) & 3) << 3;

    auto fill = [&](int buf, int k0) {
        const bf16* ga0 = A + (size_t)(mBlock + r0) * K + k0 + c0;
        const bf16* ga1 = A + (size_t)(mBlock + r1) * K + k0 + c1;
        const bf16* gb0 = Bt + (size_t)(nBlock + r0) * K + k0 + c0;
        const bf16* gb1 = Bt + (size_t)(nBlock + r1) * K + k0 + c1;
        unsigned la0 = (unsigned)(uintptr_t)&As[buf][r0][c0];
        unsigned la1 = (unsigned)(uintptr_t)&As[buf][r1][c1];
        unsigned lb0 = (unsigned)(uintptr_t)&Bs[buf][r0][c0];
        unsigned lb1 = (unsigned)(uintptr_t)&Bs[buf][r1][c1];
        asm volatile("global_load_async_to_lds_b128 %0, %1, off"
                     :: "v"(la0), "v"((unsigned long long)(uintptr_t)ga0) : "memory");
        asm volatile("global_load_async_to_lds_b128 %0, %1, off"
                     :: "v"(la1), "v"((unsigned long long)(uintptr_t)ga1) : "memory");
        asm volatile("global_load_async_to_lds_b128 %0, %1, off"
                     :: "v"(lb0), "v"((unsigned long long)(uintptr_t)gb0) : "memory");
        asm volatile("global_load_async_to_lds_b128 %0, %1, off"
                     :: "v"(lb1), "v"((unsigned long long)(uintptr_t)gb1) : "memory");
    };

    v8f z = {};
    v8f acc[2][4] = { { z, z, z, z }, { z, z, z, z } };

    // pipeline prologue: tiles 0 and 1 in flight; drain tile 0, publish it
    fill(0, 0);
    fill(1, 32);
    asm volatile("s_wait_asynccnt 0x4" ::: "memory");
    __syncthreads();

    for (int t = 0; t < nT; ++t) {
        const int cur = t % 3;
        if (t + 2 < nT) fill((t + 2) % 3, (t + 2) << 5);

        // --- fragments (ISA 16-bit layouts), vector LDS reads
        v16bf af[2], bfr[4];
        #pragma unroll
        for (int i = 0; i < 2; ++i) {
            int m = wm + i * 16 + l16;
            v8bf lo = *(const v8bf*)&As[cur][m][8 * half];       // K = 8h..8h+7
            v8bf hi = *(const v8bf*)&As[cur][m][16 + 8 * half];  // K = 16+8h..
            af[i] = __builtin_shufflevector(lo, hi, 0, 1, 2, 3, 4, 5, 6, 7,
                                            8, 9, 10, 11, 12, 13, 14, 15);
        }
        #pragma unroll
        for (int j = 0; j < 4; ++j) {
            int n = wn + j * 16 + l16;
            v8bf lo = *(const v8bf*)&Bs[cur][n][8 * half];       // K = 8h..8h+7
            v8bf hi = *(const v8bf*)&Bs[cur][n][16 + 8 * half];  // K = 16+8h..
            bfr[j] = __builtin_shufflevector(lo, hi, 0, 1, 2, 3, 4, 5, 6, 7,
                                             8, 9, 10, 11, 12, 13, 14, 15);
        }
        #pragma unroll
        for (int i = 0; i < 2; ++i)
            #pragma unroll
            for (int j = 0; j < 4; ++j)
                acc[i][j] = __builtin_amdgcn_wmma_f32_16x16x32_bf16(
                    false, af[i], false, bfr[j], (short)0, acc[i][j], false, false);

        // drain tile t+1 (keep t+2 in flight), then publish via barrier
        if (t + 2 < nT) asm volatile("s_wait_asynccnt 0x4" ::: "memory");
        else            asm volatile("s_wait_asynccnt 0x0" ::: "memory");
        __syncthreads();
    }

    // --- epilogue (C/D layout: lane<16 -> M=r, lane>=16 -> M=8+r; N=l16)
    #pragma unroll
    for (int i = 0; i < 2; ++i) {
        #pragma unroll
        for (int j = 0; j < 4; ++j) {
            int col = nBlock + wn + j * 16 + l16;
            #pragma unroll
            for (int r = 0; r < 8; ++r) {
                int row = mBlock + wm + i * 16 + half * 8 + r;
                size_t idx = (size_t)row * N + col;
                float vv = acc[i][j][r];
                if (EPI == 0) {
                    outB[idx] = (bf16)vv;
                } else if (EPI == 1) {
                    outF[idx] = res1[idx] + vv + bias[col];
                } else if (EPI == 2) {
                    float t2 = vv + bias[col];
                    outB[idx] = (bf16)(t2 > 0.f ? t2 : 0.f);
                } else {
                    outF[idx] = 0.5f * (res1[idx] + res2[idx] + vv + bias[col]);
                }
            }
        }
    }
}

// ---------------------------------------------------------------------------
// LSH hash + per-row bitonic sort. One block per (b*H+h), 256 threads.
// key = bucket*T + t (unique) -> sticker[j] = sortedkey[j] & (T-1)
// ---------------------------------------------------------------------------
__global__ __launch_bounds__(256)
void hash_sort(const bf16* __restrict__ qk, const float* __restrict__ rot,
               int* __restrict__ sticker) {
    __shared__ float rotS[DH * NB2];      // 16 KB
    __shared__ unsigned keys[TT];         // 16 KB
    const int bh = blockIdx.x;
    const int b = bh / HH, h = bh % HH;
    const int tid = threadIdx.x;

    for (int p = tid; p < DH * NB2; p += 256) rotS[p] = rot[p];
    __syncthreads();

    for (int t = tid; t < TT; t += 256) {
        float qf[DH];
        const bf16* qr = qk + ((size_t)(b * TT + t)) * DD + h * DH;
        #pragma unroll
        for (int e = 0; e < DH; ++e) qf[e] = (float)qr[e];
        float best = -3.4e38f;
        int bi = 0;
        for (int i = 0; i < NB2; ++i) {
            float r = 0.f;
            #pragma unroll
            for (int e = 0; e < DH; ++e) r += qf[e] * rotS[e * NB2 + i];
            if (r > best)  { best = r;  bi = i; }
            if (-r > best) { best = -r; bi = NB2 + i; }
        }
        keys[t] = (unsigned)bi * TT + (unsigned)t;
    }
    __syncthreads();

    // bitonic sort (ascending), 4096 elements, 2048 compare-exchanges/pass
    for (unsigned kk = 2; kk <= TT; kk <<= 1) {
        for (unsigned jj = kk >> 1; jj > 0; jj >>= 1) {
            for (unsigned p = tid; p < TT / 2; p += 256) {
                unsigned i = (p / jj) * (jj << 1) + (p % jj);
                unsigned l = i | jj;
                bool up = ((i & kk) == 0);
                unsigned a = keys[i], c = keys[l];
                if ((a > c) == up) { keys[i] = c; keys[l] = a; }
            }
            __syncthreads();
        }
    }
    for (int j = tid; j < TT; j += 256)
        sticker[(size_t)bh * TT + j] = (int)(keys[j] & (TT - 1));
}

// ---------------------------------------------------------------------------
// Bucketed attention: one block per (chunk, bh). 32 queries x 64 kv
// (look-one-back). Shared-QK: keys are unit-normalized queries.
// Scatters head output straight back to unsorted positions.
// ---------------------------------------------------------------------------
__global__ __launch_bounds__(256)
void lsh_attn(const bf16* __restrict__ qk, const bf16* __restrict__ vmat,
              const int* __restrict__ sticker, bf16* __restrict__ o) {
    __shared__ float qs[32][64];
    __shared__ float ks[64][65];
    __shared__ float vs[64][65];
    __shared__ float dots[32][64];
    __shared__ int   tkv[64];

    const int c  = blockIdx.x;
    const int bh = blockIdx.y;
    const int b = bh / HH, h = bh % HH;
    const int cp = (c + NCH - 1) % NCH;      // look-one-back chunk
    const int tid = threadIdx.x;

    if (tid < 64) {
        int chunk = (tid < 32) ? c : cp;
        tkv[tid] = sticker[(size_t)bh * TT + chunk * 32 + (tid & 31)];
    }
    __syncthreads();

    for (int p = tid; p < 64 * 64; p += 256) {
        int j = p >> 6, e = p & 63;
        size_t base = ((size_t)(b * TT + tkv[j])) * DD + h * DH + e;
        ks[j][e] = (float)qk[base];
        vs[j][e] = (float)vmat[base];
    }
    __syncthreads();
    for (int p = tid; p < 32 * 64; p += 256) {           // raw queries = chunk c rows
        int i = p >> 6, e = p & 63;
        qs[i][e] = ks[i][e];
    }
    __syncthreads();
    if (tid < 64) {                                      // normalize keys in place
        float s = 0.f;
        for (int e = 0; e < 64; ++e) s += ks[tid][e] * ks[tid][e];
        float inv = 1.f / fmaxf(sqrtf(s), 1e-12f);
        for (int e = 0; e < 64; ++e) ks[tid][e] *= inv;
    }
    __syncthreads();

    for (int p = tid; p < 32 * 64; p += 256) {
        int i = p >> 6, j = p & 63;
        float d = 0.f;
        #pragma unroll
        for (int e = 0; e < 64; ++e) d += qs[i][e] * ks[j][e];
        d *= 0.125f;                                     // D^-0.5, D=64
        if (tkv[i] == tkv[j]) d = -5e4f;                 // self-attn mask
        dots[i][j] = d;
    }
    __syncthreads();

    if (tid < 32) {                                      // softmax over 64 keys
        float mx = -3.4e38f;
        for (int j = 0; j < 64; ++j) mx = fmaxf(mx, dots[tid][j]);
        float s = 0.f;
        for (int j = 0; j < 64; ++j) { float e = __expf(dots[tid][j] - mx); dots[tid][j] = e; s += e; }
        float inv = 1.f / s;
        for (int j = 0; j < 64; ++j) dots[tid][j] *= inv;
    }
    __syncthreads();

    for (int p = tid; p < 32 * 64; p += 256) {
        int i = p >> 6, e = p & 63;
        float s = 0.f;
        #pragma unroll
        for (int j = 0; j < 64; ++j) s += dots[i][j] * vs[j][e];
        o[((size_t)(b * TT + tkv[i])) * DD + h * DH + e] = (bf16)s;   // unsort scatter
    }
}

// ---------------------------------------------------------------------------
extern "C" void kernel_launch(void* const* d_in, const int* in_sizes, int n_in,
                              void* d_out, int out_size, void* d_ws, size_t ws_size,
                              hipStream_t stream) {
    const float* x      = (const float*)d_in[0];
    const float* w_qk   = (const float*)d_in[1];
    const float* w_v    = (const float*)d_in[2];
    const float* w_out  = (const float*)d_in[3];
    const float* b_out  = (const float*)d_in[4];
    const float* ln1_g  = (const float*)d_in[5];
    const float* ln1_b  = (const float*)d_in[6];
    const float* w_ff1  = (const float*)d_in[7];
    const float* b_ff1  = (const float*)d_in[8];
    const float* w_ff2  = (const float*)d_in[9];
    const float* b_ff2  = (const float*)d_in[10];
    const float* ln2_g  = (const float*)d_in[11];
    const float* ln2_b  = (const float*)d_in[12];
    const float* rot    = (const float*)d_in[13];

    char* ws = (char*)d_ws;
    size_t off = 0;
    auto take = [&](size_t bytes) -> char* {
        char* p = ws + off;
        off = (off + bytes + 255) & ~(size_t)255;
        return p;
    };
    bf16* h_bf   = (bf16*)take((size_t)BT * DD * 2);   // LN1 / LN2 activations
    bf16* qk_bf  = (bf16*)take((size_t)BT * DD * 2);
    bf16* v_bf   = (bf16*)take((size_t)BT * DD * 2);
    bf16* o_bf   = (bf16*)take((size_t)BT * DD * 2);
    float* y1    = (float*)take((size_t)BT * DD * 4);
    bf16* mid    = (bf16*)take((size_t)BT * FF * 2);
    bf16* wqk_t  = (bf16*)take((size_t)DD * DD * 2);   // all weights stored N-major
    bf16* wv_t   = (bf16*)take((size_t)DD * DD * 2);
    bf16* wout_t = (bf16*)take((size_t)DD * DD * 2);
    bf16* wff1_t = (bf16*)take((size_t)DD * FF * 2);
    bf16* wff2_t = (bf16*)take((size_t)FF * DD * 2);
    int*  stick  = (int*)take((size_t)BB * HH * TT * 4);

    // 1) weights -> bf16, transposed to N-major (one-time)
    cvt_transpose_bf16<<<512, 256, 0, stream>>>(w_qk,  wqk_t,  DD, DD);
    cvt_transpose_bf16<<<512, 256, 0, stream>>>(w_v,   wv_t,   DD, DD);
    cvt_transpose_bf16<<<512, 256, 0, stream>>>(w_out, wout_t, DD, DD);
    cvt_transpose_bf16<<<1024, 256, 0, stream>>>(w_ff1, wff1_t, DD, FF);
    cvt_transpose_bf16<<<1024, 256, 0, stream>>>(w_ff2, wff2_t, FF, DD);

    dim3 blk(256);
    dim3 gD(DD / 128, BT / 128);    // N=512 GEMMs
    dim3 gF(FF / 128, BT / 128);    // N=2048 GEMM

    // 2) h1 = LN1(x)
    ln_to_bf16<<<BT, 256, 0, stream>>>(x, ln1_g, ln1_b, h_bf);
    // 3) qk / v projections
    gemm_bf16<0><<<gD, blk, 0, stream>>>(h_bf, wqk_t, BT, DD, DD, nullptr, qk_bf, nullptr, nullptr, nullptr);
    gemm_bf16<0><<<gD, blk, 0, stream>>>(h_bf, wv_t,  BT, DD, DD, nullptr, v_bf,  nullptr, nullptr, nullptr);
    // 4) LSH hash + sort
    hash_sort<<<BB * HH, 256, 0, stream>>>(qk_bf, rot, stick);
    // 5) bucketed attention
    lsh_attn<<<dim3(NCH, BB * HH), 256, 0, stream>>>(qk_bf, v_bf, stick, o_bf);
    // 6) y1 = x + o @ w_out + b_out
    gemm_bf16<1><<<gD, blk, 0, stream>>>(o_bf, wout_t, BT, DD, DD, y1, nullptr, b_out, x, nullptr);
    // 7) h2 = LN2(y1)
    ln_to_bf16<<<BT, 256, 0, stream>>>(y1, ln2_g, ln2_b, h_bf);
    // 8) mid = relu(h2 @ w_ff1 + b_ff1)
    gemm_bf16<2><<<gF, blk, 0, stream>>>(h_bf, wff1_t, BT, FF, DD, nullptr, mid, b_ff1, nullptr, nullptr);
    // 9) out = 0.5*(y1 + x + mid @ w_ff2 + b_ff2)
    gemm_bf16<3><<<gD, blk, 0, stream>>>(mid, wff2_t, BT, DD, FF, (float*)d_out, nullptr, b_ff2, y1, x);

    (void)in_sizes; (void)n_in; (void)out_size; (void)ws_size;
}
